// Attention_9234179687570
// MI455X (gfx1250) — compile-verified, matrix operand order checked by reading, not run on hardware
//
#include <hip/hip_runtime.h>
#include <math.h>

// ---------------------------------------------------------------------------
// CDNA5 (gfx1250) attention pipeline on V_WMMA_F32_16X16X32_BF16.
// wave32, 8 waves / 256-thread block, block tile 128x128x32, wave tile 64x32,
// double-buffered LDS with software-pipelined global->LDS staging.
// ---------------------------------------------------------------------------

typedef __attribute__((ext_vector_type(8)))  float  v8f;
typedef __attribute__((ext_vector_type(8)))  __bf16 bf16x8;
typedef __attribute__((ext_vector_type(16))) __bf16 bf16x16;

#define BM 128
#define BN 128
#define BK 32
#define BKP 40   // padded LDS stride (80 B) to de-conflict banks

// C = act(alpha * A @ B' + bias) + residual
//   A   : [M x K] f32 row-major, leading dim lda  (batched via aBatch)
//   B   : if B_IS_NK: [N x K] row-major (weights W[out,in], or K-matrix)
//         else      : [K x N] row-major (e.g. V), transposed while staging
//   res : optional residual, leading dim ldc
// Requires: M%128==0, N%128==0, K%64==0 (true for all shapes here).
template <bool B_IS_NK, bool RELU>
__global__ __launch_bounds__(256) void gemm_wmma_bf16(
    const float* __restrict__ A, const float* __restrict__ B,
    const float* __restrict__ bias, const float* __restrict__ res,
    float* __restrict__ C,
    int K, int lda, int ldb, int ldc, float alpha,
    long long aBatch, long long bBatch, long long cBatch, long long rBatch)
{
    __shared__ __bf16 As[2][BM][BKP];
    __shared__ __bf16 Bs[2][BN][BKP];

    const int tid    = threadIdx.x;
    const int lane   = tid & 31;
    const int wave   = tid >> 5;      // 0..7
    const int wave_m = wave >> 2;     // 0..1 -> 64-row slice
    const int wave_n = wave & 3;      // 0..3 -> 32-col slice

    const int m0 = blockIdx.y * BM;
    const int n0 = blockIdx.x * BN;
    const long long z = blockIdx.z;

    A += z * aBatch;
    B += z * bBatch;
    C += z * cBatch;
    const float* R = res ? res + z * rBatch : nullptr;

    v8f acc[4][2] = {};

    // staging coords: each thread owns 16 contiguous K-floats of one row
    const int arow = tid >> 1;          // 0..127
    const int kc   = (tid & 1) * 16;    // 0 or 16
    // KN-mode transpose coords: thread covers row kk, 16 consecutive cols
    const int kn_k = tid >> 3;          // 0..31
    const int kn_n = (tid & 7) * 16;    // 0,16,...,112

    // fragment coords per ISA 7.12.2 16-bit layout
    const int fr = lane & 15;
    const int kb = (lane >> 4) * 8;

    float4 ra[4], rb[4];

    auto gload = [&](int k0) {
        const float* sa = A + (long long)(m0 + arow) * lda + k0 + kc;
#pragma unroll
        for (int i = 0; i < 4; ++i) ra[i] = *(const float4*)(sa + i * 4);
        if (B_IS_NK) {
            const float* sb = B + (long long)(n0 + arow) * ldb + k0 + kc;
#pragma unroll
            for (int i = 0; i < 4; ++i) rb[i] = *(const float4*)(sb + i * 4);
        } else {
            const float* sb = B + (long long)(k0 + kn_k) * ldb + n0 + kn_n;
#pragma unroll
            for (int i = 0; i < 4; ++i) rb[i] = *(const float4*)(sb + i * 4);
        }
    };

    auto lstore = [&](int buf) {
#pragma unroll
        for (int i = 0; i < 4; ++i) {
            __bf16* d = &As[buf][arow][kc + i * 4];
            d[0] = (__bf16)ra[i].x; d[1] = (__bf16)ra[i].y;
            d[2] = (__bf16)ra[i].z; d[3] = (__bf16)ra[i].w;
        }
        if (B_IS_NK) {
#pragma unroll
            for (int i = 0; i < 4; ++i) {
                __bf16* d = &Bs[buf][arow][kc + i * 4];
                d[0] = (__bf16)rb[i].x; d[1] = (__bf16)rb[i].y;
                d[2] = (__bf16)rb[i].z; d[3] = (__bf16)rb[i].w;
            }
        } else {
#pragma unroll
            for (int i = 0; i < 4; ++i) {   // transpose: Bs[n][k]
                Bs[buf][kn_n + i * 4 + 0][kn_k] = (__bf16)rb[i].x;
                Bs[buf][kn_n + i * 4 + 1][kn_k] = (__bf16)rb[i].y;
                Bs[buf][kn_n + i * 4 + 2][kn_k] = (__bf16)rb[i].z;
                Bs[buf][kn_n + i * 4 + 3][kn_k] = (__bf16)rb[i].w;
            }
        }
    };

    auto load_frag = [&](const __bf16 (&T)[BM][BKP], int row) -> bf16x16 {
        bf16x8 lo = *(const bf16x8*)&T[row][kb];
        bf16x8 hi = *(const bf16x8*)&T[row][kb + 16];
        return __builtin_shufflevector(lo, hi, 0,1,2,3,4,5,6,7,
                                               8,9,10,11,12,13,14,15);
    };

    auto compute = [&](int buf) {
        bf16x16 af[4], bfv[2];
#pragma unroll
        for (int s = 0; s < 4; ++s)
            af[s] = load_frag(As[buf], wave_m * 64 + s * 16 + fr);
#pragma unroll
        for (int t = 0; t < 2; ++t)
            bfv[t] = load_frag(Bs[buf], wave_n * 32 + t * 16 + fr);
#pragma unroll
        for (int i = 0; i < 4; ++i)
#pragma unroll
            for (int j = 0; j < 2; ++j)
                acc[i][j] = __builtin_amdgcn_wmma_f32_16x16x32_bf16(
                    false, af[i], false, bfv[j], (short)0, acc[i][j],
                    false, false);
    };

    // prologue: stage K-slice 0 into buffer 0
    gload(0);
    lstore(0);
    __syncthreads();

    // software-pipelined main loop, unrolled x2 so buffer ids are literals
    for (int k0 = 0; k0 < K; k0 += 2 * BK) {
        gload(k0 + BK);          // prefetch next slice into registers
        compute(0);
        lstore(1);
        __syncthreads();
        if (k0 + 2 * BK < K) {
            gload(k0 + 2 * BK);
            compute(1);
            lstore(0);
            __syncthreads();
        } else {
            compute(1);
        }
    }

    // epilogue: C/D layout lane=(N%16, M-half), VGPR r = M offset
    const int fn    = lane & 15;
    const int mhalf = (lane >> 4) * 8;
#pragma unroll
    for (int i = 0; i < 4; ++i) {
#pragma unroll
        for (int j = 0; j < 2; ++j) {
            const int n = n0 + wave_n * 32 + j * 16 + fn;
            const float bv = bias ? bias[n] : 0.0f;
#pragma unroll
            for (int r = 0; r < 8; ++r) {
                const int m = m0 + wave_m * 64 + i * 16 + mhalf + r;
                float v = acc[i][j][r] * alpha + bv;
                if (RELU) v = v > 0.0f ? v : 0.0f;
                if (R) v += R[(long long)m * ldc + n];
                C[(long long)m * ldc + n] = v;
            }
        }
    }
}

// Row-wise softmax in place: one block per row of `cols` f32.
__global__ __launch_bounds__(256) void softmax_rows(float* __restrict__ s,
                                                    int cols)
{
    __shared__ float red[256];
    float* p = s + (long long)blockIdx.x * cols;
    const int tid = threadIdx.x;

    float vmax = -INFINITY;
    for (int c = tid; c < cols; c += 256) vmax = fmaxf(vmax, p[c]);
    red[tid] = vmax; __syncthreads();
    for (int w = 128; w > 0; w >>= 1) {
        if (tid < w) red[tid] = fmaxf(red[tid], red[tid + w]);
        __syncthreads();
    }
    vmax = red[0]; __syncthreads();

    float sum = 0.0f;
    for (int c = tid; c < cols; c += 256) {
        float e = __expf(p[c] - vmax);
        p[c] = e;
        sum += e;
    }
    red[tid] = sum; __syncthreads();
    for (int w = 128; w > 0; w >>= 1) {
        if (tid < w) red[tid] += red[tid + w];
        __syncthreads();
    }
    const float inv = 1.0f / red[0];
    for (int c = tid; c < cols; c += 256) p[c] *= inv;
}

extern "C" void kernel_launch(void* const* d_in, const int* in_sizes, int n_in,
                              void* d_out, int out_size, void* d_ws,
                              size_t ws_size, hipStream_t stream)
{
    (void)in_sizes; (void)n_in; (void)out_size; (void)ws_size;
    const int Bz = 4, S = 2048, H = 768;
    const int M = Bz * S;              // 8192 rows for the flat linears
    const long long SH = (long long)S * H;
    const long long SS = (long long)S * S;

    const float* query = (const float*)d_in[0];
    const float* key   = (const float*)d_in[1];
    const float* value = (const float*)d_in[2];
    const float* Wq  = (const float*)d_in[3];
    const float* bq  = (const float*)d_in[4];
    const float* Wk  = (const float*)d_in[5];
    const float* bk  = (const float*)d_in[6];
    const float* Wo1 = (const float*)d_in[7];
    const float* bo1 = (const float*)d_in[8];
    const float* Wo2 = (const float*)d_in[9];
    const float* bo2 = (const float*)d_in[10];
    float* out = (float*)d_out;

    float* ws = (float*)d_ws;
    float* Qb = ws;                       // [8192 x 768]
    float* Kb = Qb + (long long)M * H;    // [8192 x 768]
    float* Sc = Kb + (long long)M * H;    // [4 x 2048 x 2048]
    float* Wt = Sc + (long long)Bz * SS;  // [8192 x 768] weighted
    float* H1 = Qb;                       // reuse Q region (dead after scores)

    const dim3 blk(256);
    const float scale = 1.0f / sqrtf((float)H);

    // Q = query @ Wq^T + bq
    gemm_wmma_bf16<true, false><<<dim3(H / BN, M / BM, 1), blk, 0, stream>>>(
        query, Wq, bq, nullptr, Qb, H, H, H, H, 1.0f, 0, 0, 0, 0);
    // K = key @ Wk^T + bk
    gemm_wmma_bf16<true, false><<<dim3(H / BN, M / BM, 1), blk, 0, stream>>>(
        key, Wk, bk, nullptr, Kb, H, H, H, H, 1.0f, 0, 0, 0, 0);
    // scores = Q @ K^T * scale   (batched over B)
    gemm_wmma_bf16<true, false><<<dim3(S / BN, S / BM, Bz), blk, 0, stream>>>(
        Qb, Kb, nullptr, nullptr, Sc, H, H, H, S, scale, SH, SH, SS, 0);
    // softmax rows
    softmax_rows<<<dim3(Bz * S), blk, 0, stream>>>(Sc, S);
    // weighted = attn @ V + value   (batched, V is [K x N])
    gemm_wmma_bf16<false, false><<<dim3(H / BN, S / BM, Bz), blk, 0, stream>>>(
        Sc, value, nullptr, value, Wt, S, S, H, H, 1.0f, SS, SH, SH, SH);
    // h1 = relu(weighted @ Wo1^T + bo1)
    gemm_wmma_bf16<true, true><<<dim3(H / BN, M / BM, 1), blk, 0, stream>>>(
        Wt, Wo1, bo1, nullptr, H1, H, H, H, H, 1.0f, 0, 0, 0, 0);
    // out = h1 @ Wo2^T + bo2 + weighted
    gemm_wmma_bf16<true, false><<<dim3(H / BN, M / BM, 1), blk, 0, stream>>>(
        H1, Wo2, bo2, Wt, out, H, H, H, H, 1.0f, 0, 0, 0, 0);
}
